// AstrometryConcordanceHead_18863496364438
// MI455X (gfx1250) — matrix operand order repaired and checked by reading, not compiled, not run on hardware
//
#include <hip/hip_runtime.h>
#include <hip/hip_bf16.h>
#include <math.h>

typedef __attribute__((ext_vector_type(16))) _Float16 v16h;
typedef __attribute__((ext_vector_type(8)))  _Float16 v8h;
typedef __attribute__((ext_vector_type(8)))  float    v8f;

union F16V { v16h v; v8h h[2]; };
union H8   { v8h v; _Float16 e[8]; };

#define HC 128          // common grid 128x128
#define PW 132          // conv zero-padded buffer width/height (128 + 2*2)
#define RH 134          // ref replicate-padded height (128 + 2*3)
#define RW 152          // ref padded row stride (>= 134 + 16 slack for 2nd WMMA col tile)
#define DTOK 256        // token dim

// ---------- fragment loaders (per documented CDNA5 16-bit WMMA VGPR layouts) ----------
// A 16x32 (MxK): lane m=l&15, h=l>>4 holds K = h*8+{0..7} and 16+h*8+{0..7}
static __device__ __forceinline__ v16h load_afrag(const _Float16* p, int h) {
  F16V u;
  u.h[0] = *(const v8h*)(p + h * 8);
  u.h[1] = *(const v8h*)(p + 16 + h * 8);
  return u.v;
}
// B 32x16 (KxN): lane n=l&15, h=l>>4 holds K = h*8+{0..7} and 16+h*8+{0..7}
static __device__ __forceinline__ v16h load_bfrag(const _Float16* p, int h) {
  F16V u;
  const _Float16* q = p + h * 8;
  u.h[0] = *(const v8h*)(q);
  u.h[1] = *(const v8h*)(q + 16);
  return u.v;
}
static __device__ __forceinline__ v8f wmma_f16(v16h a, v16h b, v8f c) {
  return __builtin_amdgcn_wmma_f32_16x16x32_f16(false, a, false, b, (short)0, c, false, false);
}
static __device__ __forceinline__ float wave_sum32(float x) {
#pragma unroll
  for (int off = 16; off > 0; off >>= 1) x += __shfl_xor(x, off, 32);
  return x;
}
static __device__ __forceinline__ float gelu_exact(float x) {
  return 0.5f * x * (1.0f + erff(x * 0.70710678118654752f));
}

// ---------- prep: bilinear upsample rubin 64->128, L2 norm over D, f16 ----------
__global__ void prep_q_kernel(const float* __restrict__ rubin, _Float16* __restrict__ q, int B) {
  int gid = blockIdx.x * blockDim.x + threadIdx.x;
  int w = gid >> 5, lane = gid & 31;
  if (w >= B * HC * HC) return;
  int b = w / (HC * HC);
  int y = (w / HC) % HC;
  int x = w % HC;
  // half-pixel mapping, scale = 0.5
  float syf = (y + 0.5f) * 0.5f - 0.5f;
  float sxf = (x + 0.5f) * 0.5f - 0.5f;
  float fy0 = floorf(syf), fx0 = floorf(sxf);
  float wy = syf - fy0, wx = sxf - fx0;
  int iy0 = (int)fy0, ix0 = (int)fx0;
  int y0 = min(max(iy0, 0), 63), y1 = min(max(iy0 + 1, 0), 63);
  int x0 = min(max(ix0, 0), 63), x1 = min(max(ix0 + 1, 0), 63);
  int d0 = lane * 8;
  const float* p00 = rubin + ((size_t)b * 4096 + y0 * 64 + x0) * DTOK + d0;
  const float* p01 = rubin + ((size_t)b * 4096 + y0 * 64 + x1) * DTOK + d0;
  const float* p10 = rubin + ((size_t)b * 4096 + y1 * 64 + x0) * DTOK + d0;
  const float* p11 = rubin + ((size_t)b * 4096 + y1 * 64 + x1) * DTOK + d0;
  float w00 = (1.f - wy) * (1.f - wx), w01 = (1.f - wy) * wx;
  float w10 = wy * (1.f - wx), w11 = wy * wx;
  float v[8]; float ss = 0.f;
#pragma unroll
  for (int i = 0; i < 8; ++i) {
    float val = w00 * p00[i] + w01 * p01[i] + w10 * p10[i] + w11 * p11[i];
    v[i] = val; ss += val * val;
  }
  ss = wave_sum32(ss);
  float inv = 1.0f / fmaxf(sqrtf(ss), 1e-12f);
  H8 o;
#pragma unroll
  for (int i = 0; i < 8; ++i) o.e[i] = (_Float16)(v[i] * inv);
  *(v8h*)(q + (size_t)w * DTOK + d0) = o.v;
}

// ---------- prep: vis tokens L2 norm -> replicate-padded f16 ref [B][134][152][256] ----------
__global__ void prep_ref_kernel(const float* __restrict__ vis, _Float16* __restrict__ ref, int B) {
  int gid = blockIdx.x * blockDim.x + threadIdx.x;
  int w = gid >> 5, lane = gid & 31;
  if (w >= B * RH * RH) return;
  int b = w / (RH * RH);
  int r = (w / RH) % RH;
  int c = w % RH;
  int sy = min(max(r - 3, 0), HC - 1);
  int sx = min(max(c - 3, 0), HC - 1);
  int d0 = lane * 8;
  const float* p = vis + ((size_t)b * HC * HC + sy * HC + sx) * DTOK + d0;
  float v[8]; float ss = 0.f;
#pragma unroll
  for (int i = 0; i < 8; ++i) { v[i] = p[i]; ss += v[i] * v[i]; }
  ss = wave_sum32(ss);
  float inv = 1.0f / fmaxf(sqrtf(ss), 1e-12f);
  H8 o;
#pragma unroll
  for (int i = 0; i < 8; ++i) o.e[i] = (_Float16)(v[i] * inv);
  *(v8h*)(ref + (((size_t)b * RH + r) * RW + c) * DTOK + d0) = o.v;
}

// ---------- repack conv weights OIHW f32 -> [outch][tap][cin_padded] f16 ----------
__global__ void repack_w_kernel(const float* __restrict__ w0, const float* __restrict__ w1,
                                const float* __restrict__ w2,
                                _Float16* __restrict__ g0, _Float16* __restrict__ g1,
                                _Float16* __restrict__ g2) {
  int i = blockIdx.x * blockDim.x + threadIdx.x;
  const int N0 = 32 * 25 * 64, N1 = 32 * 25 * 32, N2 = 16 * 25 * 32;
  if (i < N0) {
    int c = i & 63; int tp = (i >> 6) % 25; int o = i / (25 * 64);
    float v = (c < 52) ? w0[(o * 52 + c) * 25 + tp] : 0.f;
    g0[(o * 25 + tp) * 64 + c] = (_Float16)v;
  } else if ((i -= N0) < N1) {
    int c = i & 31; int tp = (i >> 5) % 25; int o = i / (25 * 32);
    g1[(o * 25 + tp) * 32 + c] = (_Float16)w1[(o * 32 + c) * 25 + tp];
  } else if ((i -= N1) < N2) {
    int c = i & 31; int tp = (i >> 5) % 25; int o = i / (25 * 32);
    float v = (o < 2) ? w2[(o * 32 + c) * 25 + tp] : 0.f;
    g2[(o * 25 + tp) * 32 + c] = (_Float16)v;
  }
}

__global__ void zero_u32_kernel(unsigned int* __restrict__ p, long long n) {
  long long i = (long long)blockIdx.x * blockDim.x + threadIdx.x;
  if (i < n) p[i] = 0u;
}

// ---------- correlation volume + soft-argmax via WMMA diagonal-band extraction ----------
// one wave per (b, y, 16-wide x tile). C(i,j) = q[x0+i] . ref_pad[y+dy][x0+j]; need j = i+dx+3.
__global__ void corr_kernel(const _Float16* __restrict__ q, const _Float16* __restrict__ ref,
                            const float* __restrict__ log_temp,
                            _Float16* __restrict__ xin, float* __restrict__ raw, int B) {
  __shared__ float cf[4][7][16][33];   // [wave][dy][row i][col j], stride 33 avoids conflicts
  int wid = threadIdx.x >> 5, lane = threadIdx.x & 31;
  int t = blockIdx.x * 4 + wid;
  int b = t >> 10;
  int y = (t >> 3) & 127;
  int x0 = (t & 7) << 4;
  int n = lane & 15, h = lane >> 4;

  // preload A (query) fragments for all 8 K-chunks (K = 256)
  const _Float16* qp = q + (((size_t)b * HC + y) * HC + x0 + n) * DTOK;
  v16h A[8];
#pragma unroll
  for (int k = 0; k < 8; ++k) A[k] = load_afrag(qp + k * 32, h);

  for (int dy = 0; dy < 7; ++dy) {
    const _Float16* rrow = ref + ((size_t)(b * RH + y + dy) * RW) * DTOK;
    const _Float16* bp0 = rrow + (size_t)(x0 + n) * DTOK;
    const _Float16* bp1 = bp0 + (size_t)16 * DTOK;
    v8f acc0 = {0,0,0,0,0,0,0,0};
    v8f acc1 = {0,0,0,0,0,0,0,0};
#pragma unroll
    for (int k = 0; k < 8; ++k) {
      v16h bf0 = load_bfrag(bp0 + k * 32, h);
      v16h bf1 = load_bfrag(bp1 + k * 32, h);
      acc0 = wmma_f16(A[k], bf0, acc0);
      acc1 = wmma_f16(A[k], bf1, acc1);
    }
#pragma unroll
    for (int v = 0; v < 8; ++v) {
      int M = h * 8 + v;                  // C row per documented f32 C/D layout
      cf[wid][dy][M][n]      = acc0[v];
      cf[wid][dy][M][16 + n] = acc1[v];
    }
  }
  asm volatile("s_wait_dscnt 0" ::: "memory");   // LDS writes visible before band gather

  if (lane < 16) {
    int i = lane, x = x0 + i;
    _Float16* xp = xin + (((size_t)b * PW + (y + 2)) * PW + (x + 2)) * 64;
    float temp = expf(log_temp[0]);
    float mx = -1e30f;
    for (int s = 0; s < 49; ++s) {
      int dyi = s / 7, dxi = s % 7;
      float c = cf[wid][dyi][i][i + dxi + 3];
      xp[2 + s] = (_Float16)c;
      mx = fmaxf(mx, c);
    }
    float Z = 0.f, sy = 0.f, sx = 0.f;
    for (int s = 0; s < 49; ++s) {
      int dyi = s / 7, dxi = s % 7;
      float c = cf[wid][dyi][i][i + dxi + 3];
      float wgt = expf((c - mx) / temp);
      Z += wgt; sy += wgt * (float)(dyi - 3); sx += wgt * (float)(dxi - 3);
    }
    float inv = 1.0f / Z;
    float rdy = sy * inv, rdx = sx * inv, conf = inv;  // max weight = exp(0)/Z
    xp[0]  = (_Float16)rdy;
    xp[1]  = (_Float16)rdx;
    xp[51] = (_Float16)conf;
#pragma unroll
    for (int cch = 52; cch < 64; ++cch) xp[cch] = (_Float16)0.f;
    float* rp = raw + (((size_t)b * HC + y) * HC + x) * 3;
    rp[0] = rdy; rp[1] = rdx; rp[2] = conf;
  }
}

// ---------- 5x5 conv (implicit GEMM, 32 out ch) + bias + exact GELU ----------
// CIN is compile-time so all strides fold into immediate load offsets.
template <int CIN>
__global__ void conv_gelu_kernel(const _Float16* __restrict__ src,
                                 const _Float16* __restrict__ wgt,
                                 const float* __restrict__ bias,
                                 _Float16* __restrict__ dst, int B) {
  int wid = threadIdx.x >> 5, lane = threadIdx.x & 31;
  int t = blockIdx.x * 4 + wid;
  int b = t >> 10;
  int y = (t >> 3) & 127;
  int x0 = (t & 7) << 4;
  int n = lane & 15, h = lane >> 4;
  const _Float16* wm0 = wgt + (size_t)n * 25 * CIN;          // A rows: out ch 0..15
  const _Float16* wm1 = wgt + (size_t)(n + 16) * 25 * CIN;   // A rows: out ch 16..31
  v8f acc0 = {0,0,0,0,0,0,0,0};
  v8f acc1 = {0,0,0,0,0,0,0,0};
  const _Float16* srow = src + (((size_t)b * PW + y) * PW + (x0 + n)) * CIN;
#pragma unroll
  for (int ty = 0; ty < 5; ++ty) {
#pragma unroll
    for (int tx = 0; tx < 5; ++tx) {
      int tap = ty * 5 + tx;
      const _Float16* bp = srow + ((size_t)ty * PW + tx) * CIN;
      const _Float16* ap0 = wm0 + tap * CIN;
      const _Float16* ap1 = wm1 + tap * CIN;
#pragma unroll
      for (int c0 = 0; c0 < CIN; c0 += 32) {
        v16h bf = load_bfrag(bp + c0, h);
        v16h a0 = load_afrag(ap0 + c0, h);
        acc0 = wmma_f16(a0, bf, acc0);
        v16h a1 = load_afrag(ap1 + c0, h);
        acc1 = wmma_f16(a1, bf, acc1);
      }
    }
  }
  _Float16* dp = dst + (((size_t)b * PW + (y + 2)) * PW + (x0 + n + 2)) * 32;
  H8 s0, s1;
#pragma unroll
  for (int v = 0; v < 8; ++v) {
    int ch0 = h * 8 + v;
    int ch1 = 16 + h * 8 + v;
    s0.e[v] = (_Float16)gelu_exact(acc0[v] + bias[ch0]);
    s1.e[v] = (_Float16)gelu_exact(acc1[v] + bias[ch1]);
  }
  *(v8h*)(dp + h * 8)      = s0.v;
  *(v8h*)(dp + 16 + h * 8) = s1.v;
}

// ---------- conv2 (2 out ch padded to 16) + residual add + output assembly ----------
__global__ void conv2_final_kernel(const _Float16* __restrict__ src,
                                   const _Float16* __restrict__ wgt,
                                   const float* __restrict__ bias2,
                                   const float* __restrict__ raw,
                                   const int* __restrict__ img_h,
                                   const int* __restrict__ img_w,
                                   float* __restrict__ out, int B) {
  int wid = threadIdx.x >> 5, lane = threadIdx.x & 31;
  int t = blockIdx.x * 4 + wid;
  int b = t >> 10;
  int y = (t >> 3) & 127;
  int x0 = (t & 7) << 4;
  int n = lane & 15, h = lane >> 4;
  const _Float16* wm = wgt + (size_t)n * 25 * 32;
  v8f acc = {0,0,0,0,0,0,0,0};
  const _Float16* srow = src + (((size_t)b * PW + y) * PW + (x0 + n)) * 32;
#pragma unroll
  for (int ty = 0; ty < 5; ++ty) {
#pragma unroll
    for (int tx = 0; tx < 5; ++tx) {
      int tap = ty * 5 + tx;
      const _Float16* bp = srow + ((size_t)ty * PW + tx) * 32;
      v16h bf = load_bfrag(bp, h);
      v16h a = load_afrag(wm + tap * 32, h);
      acc = wmma_f16(a, bf, acc);
    }
  }
  if (h == 0) {                      // lanes 0..15 hold C rows 0..7 for pixel n
    int x = x0 + n;
    float r0 = acc[0] + bias2[0];    // residual dy
    float r1 = acc[1] + bias2[1];    // residual dx
    const float* rp = raw + (((size_t)b * HC + y) * HC + x) * 3;
    float rdy = rp[0], rdx = rp[1], conf = rp[2];
    float fy = rdy + r0, fx = rdx + r1;
    float sy = (float)img_h[0] * 0.1f / 128.0f;
    float sx = (float)img_w[0] * 0.1f / 128.0f;
    size_t base = (size_t)b * 5 * HC * HC + (size_t)y * HC + x;
    out[base + 0 * HC * HC] = fx * sx;   // dra
    out[base + 1 * HC * HC] = fy * sy;   // ddec
    out[base + 2 * HC * HC] = rdy;
    out[base + 3 * HC * HC] = rdx;
    out[base + 4 * HC * HC] = conf;
  }
}

extern "C" void kernel_launch(void* const* d_in, const int* in_sizes, int n_in,
                              void* d_out, int out_size, void* d_ws, size_t ws_size,
                              hipStream_t stream) {
  const float* rubin    = (const float*)d_in[0];
  const float* vis      = (const float*)d_in[1];
  const float* w0       = (const float*)d_in[2];
  const float* b0       = (const float*)d_in[3];
  const float* w1       = (const float*)d_in[4];
  const float* b1       = (const float*)d_in[5];
  const float* w2       = (const float*)d_in[6];
  const float* b2       = (const float*)d_in[7];
  const float* log_temp = (const float*)d_in[8];
  const int*   img_h    = (const int*)d_in[13];
  const int*   img_w    = (const int*)d_in[14];
  int B = in_sizes[0] / (64 * 64 * 256);

  char* ws = (char*)d_ws;
  size_t off = 0;
  auto wsalloc = [&](size_t bytes) -> size_t {
    size_t o = off; off += (bytes + 255) & ~(size_t)255; return o;
  };
  size_t o_q   = wsalloc((size_t)B * HC * HC * DTOK * 2);
  size_t o_ref = wsalloc((size_t)B * RH * RW * DTOK * 2);
  size_t o_xin = wsalloc((size_t)B * PW * PW * 64 * 2);
  size_t o_raw = wsalloc((size_t)B * HC * HC * 3 * 4);
  size_t o_w0g = wsalloc(32 * 25 * 64 * 2);
  size_t o_w1g = wsalloc(32 * 25 * 32 * 2);
  size_t o_w2g = wsalloc(16 * 25 * 32 * 2);
  // h0/h1 alias the q region: q is dead after corr_kernel completes (stream-ordered)
  size_t hbytes = ((size_t)B * PW * PW * 32 * 2 + 255) & ~(size_t)255;
  size_t o_h0 = o_q;
  size_t o_h1 = o_q + hbytes;

  _Float16* q    = (_Float16*)(ws + o_q);
  _Float16* ref  = (_Float16*)(ws + o_ref);
  _Float16* xin  = (_Float16*)(ws + o_xin);
  float*    raw  = (float*)(ws + o_raw);
  _Float16* g0   = (_Float16*)(ws + o_w0g);
  _Float16* g1   = (_Float16*)(ws + o_w1g);
  _Float16* g2   = (_Float16*)(ws + o_w2g);
  _Float16* h0   = (_Float16*)(ws + o_h0);
  _Float16* h1   = (_Float16*)(ws + o_h1);

  int waves_q = B * HC * HC;
  prep_q_kernel<<<(waves_q * 32 + 255) / 256, 256, 0, stream>>>(rubin, q, B);
  int waves_r = B * RH * RH;
  prep_ref_kernel<<<(waves_r * 32 + 255) / 256, 256, 0, stream>>>(vis, ref, B);
  repack_w_kernel<<<(32*25*64 + 32*25*32 + 16*25*32 + 255) / 256, 256, 0, stream>>>(
      w0, w1, w2, g0, g1, g2);

  long long nz_xin = (long long)B * PW * PW * 64 / 2;   // u32 words
  zero_u32_kernel<<<(int)((nz_xin + 255) / 256), 256, 0, stream>>>(
      (unsigned int*)(ws + o_xin), nz_xin);

  corr_kernel<<<B * 256, 128, 0, stream>>>(q, ref, log_temp, xin, raw, B);

  long long nz_h = (long long)B * PW * PW * 32 / 2;
  zero_u32_kernel<<<(int)((nz_h + 255) / 256), 256, 0, stream>>>((unsigned int*)h0, nz_h);
  zero_u32_kernel<<<(int)((nz_h + 255) / 256), 256, 0, stream>>>((unsigned int*)h1, nz_h);

  conv_gelu_kernel<64><<<B * 256, 128, 0, stream>>>(xin, g0, b0, h0, B);
  conv_gelu_kernel<32><<<B * 256, 128, 0, stream>>>(h0, g1, b1, h1, B);
  conv2_final_kernel<<<B * 256, 128, 0, stream>>>(h1, g2, b2, raw, img_h, img_w,
                                                  (float*)d_out, B);
  (void)n_in; (void)out_size; (void)ws_size;
}
